// MultiHeadAttention1_58437325030128
// MI455X (gfx1250) — matrix operand order repaired and checked by reading, not compiled, not run on hardware
//
#include <hip/hip_runtime.h>
#include <hip/hip_bf16.h>
#include <math.h>

typedef __bf16 bf16;
typedef __attribute__((ext_vector_type(16))) __bf16 v16bf;
typedef __attribute__((ext_vector_type(8)))  __bf16 v8bf;
typedef __attribute__((ext_vector_type(4)))  __bf16 v4bf;
typedef __attribute__((ext_vector_type(8)))  float  v8f;

#define HIDDEN 256
#define HEADS  8
#define ATT    32
#define TE_DIM 64
#define SEQ    10
#define DIN    320      // HIDDEN + TE_DIM
#define NQKV   768      // 3*HIDDEN  (Q|K|V columns)

#define NB     8        // batch elements per workgroup
#define ROWS   80       // NB*SEQ  (exactly 5 WMMA M-tiles)
#define MT     5
#define ROWSP  96       // padded row count for QK tail
#define XS     336      // X LDS col stride (320+16)
#define QKS    528      // Q|K LDS col stride (512+16)
#define VSLOT  136      // V: per-dim slot stride (NB*16 + 8)
#define OS     272      // attention-out LDS col stride (256+16)
#define OUTS   264      // staged f32 output col stride (256+8)

// workspace element counts (bf16 unless noted)
#define WQKVT_E (NQKV*DIN)       // 245760
#define WOT_E   (HIDDEN*HIDDEN)  // 65536
#define TE_E    (SEQ*TE_DIM)     // 640

// LDS layout (bytes)
#define XS_BYTES  (ROWS*XS*2)        // 53760
#define QK_BYTES  (ROWSP*QKS*2)      // 101376 (>= 80*264*4 = 84480 for OutS reuse)
#define VT_BYTES  (HIDDEN*VSLOT*2)   // 69632
#define SC_BYTES  (8*16*17*4)        // 8704  (per-wave 16x17 f32 score buf)
#define AB_BYTES  (8*16*32*2)        // 8192  (per-wave 16x32 bf16 attn buf)
#define SMEM_BYTES (XS_BYTES+QK_BYTES+VT_BYTES+SC_BYTES+AB_BYTES)  // 241664

static __device__ __forceinline__ v8f wmma_bf16(v16bf a, v16bf b, v8f c) {
  return __builtin_amdgcn_wmma_f32_16x16x32_bf16(false, a, false, b,
                                                 (short)0, c, false, false);
}

static __device__ __forceinline__ v16bf cat16(v8bf lo, v8bf hi) {
  return __builtin_shufflevector(lo, hi, 0,1,2,3,4,5,6,7,8,9,10,11,12,13,14,15);
}

// ---------------- prep: weights -> bf16 transposed, te table, packed bias ---
__global__ void prep_kernel(const float* __restrict__ Wq,
                            const float* __restrict__ Wk,
                            const float* __restrict__ Wv,
                            const float* __restrict__ Wo,
                            const float* __restrict__ te_w,
                            const float* __restrict__ bq,
                            const float* __restrict__ bk,
                            const float* __restrict__ bv,
                            bf16* __restrict__ wqkvT,
                            bf16* __restrict__ woT,
                            bf16* __restrict__ teB,
                            float* __restrict__ bqkv) {
  int i = blockIdx.x * blockDim.x + threadIdx.x;
  if (i < WQKVT_E) {
    int n = i / DIN, k = i % DIN;                 // wqkvT[n][k] = W[k][n%256]
    const float* W = (n < 256) ? Wq : ((n < 512) ? Wk : Wv);
    int c = n & 255;
    wqkvT[n * DIN + k] = (bf16)W[k * HIDDEN + c];
  } else if (i < WQKVT_E + WOT_E) {
    int j = i - WQKVT_E;
    int n = j / HIDDEN, k = j % HIDDEN;
    woT[n * HIDDEN + k] = (bf16)Wo[k * HIDDEN + n];
  } else if (i < WQKVT_E + WOT_E + TE_E) {
    int j = i - WQKVT_E - WOT_E;
    int s = j / TE_DIM, c = j % TE_DIM;
    float t = (float)(SEQ - s);                   // t = 10 - s
    teB[s * TE_DIM + c] = (bf16)cosf(logf(t + 1.0f) * te_w[c]);
  } else if (i < WQKVT_E + WOT_E + TE_E + NQKV) {
    int n = i - WQKVT_E - WOT_E - TE_E;
    bqkv[n] = (n < 256) ? bq[n] : ((n < 512) ? bk[n - 256] : bv[n - 512]);
  }
}

// ---------------- fused MHA ------------------------------------------------
__global__ __launch_bounds__(256)
void mha_fused(const float* __restrict__ batch,
               const bf16* __restrict__ wqkvT,
               const bf16* __restrict__ woT,
               const bf16* __restrict__ teB,
               const float* __restrict__ bqkv,
               const float* __restrict__ bo,
               float* __restrict__ out,
               long totalRows) {
  extern __shared__ char smem[];
  bf16*  Xs = (bf16*)smem;                                       // [80][336]
  bf16*  QK = (bf16*)(smem + XS_BYTES);                          // [96][528]
  bf16*  VT = (bf16*)(smem + XS_BYTES + QK_BYTES);               // [256][136]
  float* SC = (float*)(smem + XS_BYTES + QK_BYTES + VT_BYTES);   // [8][16*17]
  bf16*  AB = (bf16*)(smem + XS_BYTES + QK_BYTES + VT_BYTES + SC_BYTES); // [8][16*32]
  bf16*  Os   = Xs;            // reuse X region: attention output [80][272] bf16
  float* OutS = (float*)QK;    // reuse QK region: final f32 tile  [80][264]

  const int tid  = threadIdx.x;
  const int wave = tid >> 5;
  const int lane = tid & 31;
  const int l16  = lane & 15;
  const int hh   = lane >> 4;
  const long base_row = (long)blockIdx.x * ROWS;

  // ---- Phase 1: stage X = [batch | te] as bf16 ----
  for (int it = tid; it < ROWS * (HIDDEN / 4); it += 256) {
    int r = it >> 6, c4 = it & 63;
    long gr = base_row + r;
    float4 v = {0.f, 0.f, 0.f, 0.f};
    if (gr < totalRows) v = ((const float4*)(batch + gr * HIDDEN))[c4];
    v4bf p; p[0] = (bf16)v.x; p[1] = (bf16)v.y; p[2] = (bf16)v.z; p[3] = (bf16)v.w;
    *(v4bf*)(Xs + r * XS + c4 * 4) = p;
  }
  for (int it = tid; it < ROWS * TE_DIM; it += 256) {
    int r = it >> 6, c = it & 63;
    Xs[r * XS + (DIN - TE_DIM) + c] = teB[(r % SEQ) * TE_DIM + c];
  }
  for (int it = tid; it < 16 * QKS; it += 256)             // zero QK tail rows
    QK[(ROWS + it / QKS) * QKS + (it % QKS)] = (bf16)0.f;
  for (int it = tid; it < (HIDDEN * VSLOT) / 2; it += 256) // zero all V slots
    ((unsigned int*)VT)[it] = 0u;
  __syncthreads();

  // ---- Phase 2: QKV = X @ WqkvT^T + b ; 5 x 48 tiles of 16x16 ----
  for (int tile = wave; tile < MT * (NQKV / 16); tile += 8) {
    int mt = tile / (NQKV / 16);
    int nt = tile % (NQKV / 16);
    int col = nt * 16 + l16;
    float bias = bqkv[col];
    v8f acc;
#pragma unroll
    for (int r = 0; r < 8; ++r) acc[r] = bias;
#pragma unroll
    for (int kt = 0; kt < DIN / 32; ++kt) {
      int k0 = kt * 32;
      int arow = mt * 16 + l16;
      v8bf alo = *(const v8bf*)(Xs + arow * XS + k0 + 16 * hh);
      v8bf ahi = *(const v8bf*)(Xs + arow * XS + k0 + 16 * hh + 8);
      v16bf a = cat16(alo, ahi);
      v16bf b = *(const v16bf*)(wqkvT + (long)col * DIN + k0 + 16 * hh);
      acc = wmma_bf16(a, b, acc);
    }
    if (col < 512) {                       // Q|K row-major
#pragma unroll
      for (int r = 0; r < 8; ++r)
        QK[(mt * 16 + 8 * hh + r) * QKS + col] = (bf16)acc[r];
    } else {                               // V dim-major, 16-row batch slots
      int c = col - 512;
#pragma unroll
      for (int r = 0; r < 8; ++r) {
        int row = mt * 16 + 8 * hh + r;
        if (row < ROWS)
          VT[c * VSLOT + (row / SEQ) * 16 + (row % SEQ)] = (bf16)acc[r];
      }
    }
  }
  __syncthreads();

  // ---- Phase 3: per-(batch,head) attention; 64 pairs over 8 waves ----
  const float scl = 0.17677669529663687f;  // 1/sqrt(32)
  float* sb = SC + wave * (16 * 17);
  bf16*  ab = AB + wave * (16 * 32);
  // zero attn buffer once; only cols 0..9 of rows 0..9 are rewritten per pair
  for (int it = lane; it < 16 * 32; it += 32) ab[it] = (bf16)0.f;
  for (int pair = wave; pair < NB * HEADS; pair += 8) {
    int b  = pair >> 3;
    int h  = pair & 7;
    int r0 = b * SEQ;
    int d0 = h * ATT;
    // scores = (Q 16x32) x (K^T 32x16)
    v8bf qlo = *(const v8bf*)(QK + (r0 + l16) * QKS + d0 + 16 * hh);
    v8bf qhi = *(const v8bf*)(QK + (r0 + l16) * QKS + d0 + 16 * hh + 8);
    v16bf qa = cat16(qlo, qhi);
    v16bf kb = *(const v16bf*)(QK + (r0 + l16) * QKS + 256 + d0 + 16 * hh);
    v8f sc0;
#pragma unroll
    for (int r = 0; r < 8; ++r) sc0[r] = 0.f;
    sc0 = wmma_bf16(qa, kb, sc0);
#pragma unroll
    for (int r = 0; r < 8; ++r)
      sb[(8 * hh + r) * 17 + l16] = sc0[r] * scl;
    asm volatile("s_wait_dscnt 0" ::: "memory");
    if (lane < SEQ) {                      // lane = query row, softmax over 10 keys
      float vals[SEQ];
      float mx = -1e30f;
#pragma unroll
      for (int k = 0; k < SEQ; ++k) {
        float v = sb[lane * 17 + k];
        v = (v == 0.f) ? -9.0e15f : v;     // reference's zero-score mask
        vals[k] = v;
        mx = fmaxf(mx, v);
      }
      float sum = 0.f;
#pragma unroll
      for (int k = 0; k < SEQ; ++k) { float e = __expf(vals[k] - mx); vals[k] = e; sum += e; }
      float inv = 1.f / sum;
#pragma unroll
      for (int k = 0; k < SEQ; ++k) ab[lane * 32 + k] = (bf16)(vals[k] * inv);
    }
    asm volatile("s_wait_dscnt 0" ::: "memory");
    // out = (attn 16x32) x (V 32x16) twice (dims 0..15, 16..31)
    v8bf plo = *(const v8bf*)(ab + l16 * 32 + 16 * hh);
    v8bf phi = *(const v8bf*)(ab + l16 * 32 + 16 * hh + 8);
    v16bf pa = cat16(plo, phi);
    v16bf vb0 = *(const v16bf*)(VT + (d0 + l16) * VSLOT + b * 16 + 16 * hh);
    v16bf vb1 = *(const v16bf*)(VT + (d0 + 16 + l16) * VSLOT + b * 16 + 16 * hh);
    v8f o0, o1;
#pragma unroll
    for (int r = 0; r < 8; ++r) { o0[r] = 0.f; o1[r] = 0.f; }
    o0 = wmma_bf16(pa, vb0, o0);
    o1 = wmma_bf16(pa, vb1, o1);
#pragma unroll
    for (int r = 0; r < 8; ++r) {
      int q = 8 * hh + r;
      if (q < SEQ) {
        Os[(r0 + q) * OS + d0 + l16]      = (bf16)o0[r];
        Os[(r0 + q) * OS + d0 + 16 + l16] = (bf16)o1[r];
      }
    }
  }
  __syncthreads();   // all phase-3 QK/VT reads done; QK region becomes OutS

  // ---- Phase 4: out = O @ Wo^T + bo ; 5 x 16 tiles, staged in LDS ----
  for (int tile = wave; tile < MT * (HIDDEN / 16); tile += 8) {
    int mt = tile / (HIDDEN / 16);
    int nt = tile % (HIDDEN / 16);
    int col = nt * 16 + l16;
    float bias = bo[col];
    v8f acc;
#pragma unroll
    for (int r = 0; r < 8; ++r) acc[r] = bias;
#pragma unroll
    for (int kt = 0; kt < HIDDEN / 32; ++kt) {
      int k0 = kt * 32;
      int arow = mt * 16 + l16;
      v8bf alo = *(const v8bf*)(Os + arow * OS + k0 + 16 * hh);
      v8bf ahi = *(const v8bf*)(Os + arow * OS + k0 + 16 * hh + 8);
      v16bf a = cat16(alo, ahi);
      v16bf bm = *(const v16bf*)(woT + (long)col * HIDDEN + k0 + 16 * hh);
      acc = wmma_bf16(a, bm, acc);
    }
#pragma unroll
    for (int r = 0; r < 8; ++r)
      OutS[(mt * 16 + 8 * hh + r) * OUTS + col] = acc[r];
  }
  __syncthreads();

  // ---- Phase 5: fully-coalesced f32 store (b128 per lane, dense stream) ----
  for (int it = tid; it < ROWS * (HIDDEN / 4); it += 256) {
    int r = it >> 6, c4 = it & 63;
    long gr = base_row + r;
    if (gr < totalRows) {
      float4 v = ((const float4*)(OutS + r * OUTS))[c4];
      ((float4*)(out + gr * HIDDEN))[c4] = v;
    }
  }
}

extern "C" void kernel_launch(void* const* d_in, const int* in_sizes, int n_in,
                              void* d_out, int out_size, void* d_ws, size_t ws_size,
                              hipStream_t stream) {
  const float* batch = (const float*)d_in[0];
  const float* Wq    = (const float*)d_in[1];
  const float* bq    = (const float*)d_in[2];
  const float* Wk    = (const float*)d_in[3];
  const float* bk    = (const float*)d_in[4];
  const float* Wv    = (const float*)d_in[5];
  const float* bv    = (const float*)d_in[6];
  const float* Wo    = (const float*)d_in[7];
  const float* bo    = (const float*)d_in[8];
  const float* te_w  = (const float*)d_in[9];
  float* out = (float*)d_out;

  bf16*  wqkvT = (bf16*)d_ws;
  bf16*  woT   = wqkvT + WQKVT_E;
  bf16*  teB   = woT + WOT_E;
  float* bqkv  = (float*)(teB + TE_E);   // byte offset 623872, 4B aligned

  long totalRows = (long)(in_sizes[0] / HIDDEN);   // B*SEQ
  int  blocks    = (int)((totalRows + ROWS - 1) / ROWS);

  int prepTotal = WQKVT_E + WOT_E + TE_E + NQKV;
  prep_kernel<<<(prepTotal + 255) / 256, 256, 0, stream>>>(
      Wq, Wk, Wv, Wo, te_w, bq, bk, bv, wqkvT, woT, teB, bqkv);

  (void)hipFuncSetAttribute((const void*)mha_fused,
                            hipFuncAttributeMaxDynamicSharedMemorySize,
                            SMEM_BYTES);
  mha_fused<<<blocks, 256, SMEM_BYTES, stream>>>(
      batch, wqkvT, woT, teB, bqkv, bo, out, totalRows);
}
